// SelfAttention_39075612459744
// MI455X (gfx1250) — matrix operand order repaired and checked by reading, not compiled, not run on hardware
//
#include <hip/hip_runtime.h>
#include <hip/hip_bf16.h>
#include <stdint.h>

// ---------------------------------------------------------------------------
// Self-attention (B=8, C_IN=256, N=4096, D_KQ=64, D_OUT=256) for gfx1250.
// ~95 GFLOP vs ~70 MB traffic -> compute bound -> bf16 WMMA flash attention,
// with CDNA5 async global->LDS staging (ASYNCcnt) and double-buffered tiles.
// ---------------------------------------------------------------------------

#define BATCH 8
#define CIN   256
#define SEQ   4096
#define DKQ   64
#define DOUT  256

typedef __attribute__((ext_vector_type(16))) __bf16 v16bf;
typedef __attribute__((ext_vector_type(8)))  float  v8f;

// ---------------------------------------------------------------------------
// Kernel 1: fused projections  q = (x^T Wq^T)/8, k = x^T Wk^T, v^T = Wv x
// One wave computes one 16(n) x 16(d) tile with WMMA bf16, K-loop over C=256.
//   Q : [b][n][64]  bf16 (pre-scaled by 1/sqrt(64))
//   K : [b][n][64]  bf16
//   VT: [b][d][N]   bf16 (transposed so attention B-fragments are contiguous)
// ---------------------------------------------------------------------------
__global__ void proj_kernel(const float* __restrict__ x,
                            const float* __restrict__ Wq,
                            const float* __restrict__ Wk,
                            const float* __restrict__ Wv,
                            __bf16* __restrict__ Q,
                            __bf16* __restrict__ K,
                            __bf16* __restrict__ VT) {
    const int wave = (blockIdx.x * blockDim.x + threadIdx.x) >> 5;
    const int lane = threadIdx.x & 31;
    const int hi = lane >> 4;      // lane half (0: lanes 0-15, 1: lanes 16-31)
    const int lo = lane & 15;

    // tile decomposition: 24 d-tiles (4 Q, 4 K, 16 V) x 256 n-tiles x 8 batches
    const int dt = wave % 24;
    const int nt = (wave / 24) % (SEQ / 16);
    const int b  = wave / (24 * (SEQ / 16));
    const int n0 = nt * 16;

    const float* W;
    int d0;
    if (dt < 4)      { W = Wq; d0 = dt * 16; }
    else if (dt < 8) { W = Wk; d0 = (dt - 4) * 16; }
    else             { W = Wv; d0 = (dt - 8) * 16; }

    const float* xb = x + (size_t)b * CIN * SEQ;

    v8f acc = {};
    for (int c0 = 0; c0 < CIN; c0 += 32) {
        // A fragment (16n x 32c): lane holds row M=lo; halfs 0-7 -> K=8*hi+h,
        // halfs 8-15 -> K=16+8*hi+(h-8)  (ISA 16-bit A 16x32 layout)
        v16bf a;
        const int row = n0 + lo;
#pragma unroll
        for (int h = 0; h < 16; ++h) {
            const int k = (h < 8 ? h : h + 8) + 8 * hi;
            a[h] = (__bf16)xb[(size_t)(c0 + k) * SEQ + row];
        }
        // B fragment (32c x 16d): lane holds col N=lo; same K order as A
        v16bf bf;
        const float* wrow = W + (size_t)(d0 + lo) * CIN + c0 + 8 * hi;
#pragma unroll
        for (int h = 0; h < 16; ++h) bf[h] = (__bf16)wrow[h < 8 ? h : h + 8];

        acc = __builtin_amdgcn_wmma_f32_16x16x32_bf16(false, a, false, bf,
                                                      (short)0, acc, false, false);
    }

    // C/D layout: lane holds col N=lo, rows M = r + 8*hi in VGPR r
    if (dt < 8) {
        __bf16* out = (dt < 4 ? Q : K) + (size_t)b * SEQ * DKQ;
        const float s = (dt < 4) ? 0.125f : 1.0f;   // 1/sqrt(D_KQ) folded into Q
#pragma unroll
        for (int r = 0; r < 8; ++r) {
            const int m = r + 8 * hi;
            out[(size_t)(n0 + m) * DKQ + d0 + lo] = (__bf16)(acc[r] * s);
        }
    } else {
        __bf16* out = VT + (size_t)b * DOUT * SEQ;
#pragma unroll
        for (int r = 0; r < 8; ++r) {
            const int m = r + 8 * hi;
            out[(size_t)(d0 + lo) * SEQ + n0 + m] = (__bf16)acc[r];
        }
    }
}

// ---------------------------------------------------------------------------
// Kernel 2: flash attention. 4 wave32s / block; each wave owns 16 query rows
// (block = 64 rows). 32-key tiles double-buffered in LDS via async loads.
// ---------------------------------------------------------------------------
#define SK_STRIDE 72   // halfs per row of K tile   (32 x 64, padded, 16B-mult)
#define SV_STRIDE 40   // halfs per row of V^T tile (256 x 32, padded)
#define SP_STRIDE 40   // halfs per row of P tile   (16 x 32, padded)
#define SO_STRIDE 68   // floats per row of O stage (64 x 64, padded)

#define SK_BYTES  (32 * SK_STRIDE * 2)        // 4608
#define SV_BYTES  (DOUT * SV_STRIDE * 2)      // 20480
#define BUF_BYTES (SK_BYTES + SV_BYTES)       // 25088 per pipeline stage
#define SMEM_P    (2 * BUF_BYTES)             // 50176
#define SMEM_SZ   (SMEM_P + 4 * 16 * SP_STRIDE * 2)   // 55296 (>= O stage 17408)

__global__ void attn_kernel(const __bf16* __restrict__ Q,
                            const __bf16* __restrict__ K,
                            const __bf16* __restrict__ VT,
                            float* __restrict__ out) {
    __shared__ __align__(16) unsigned char smem[SMEM_SZ];

    const int b     = blockIdx.x / (SEQ / 64);
    const int n0blk = (blockIdx.x % (SEQ / 64)) * 64;
    const int wid   = threadIdx.x >> 5;
    const int lane  = threadIdx.x & 31;
    const int hi    = lane >> 4;
    const int lo    = lane & 15;
    const int n0    = n0blk + wid * 16;

    __bf16* sP = (__bf16*)(smem + SMEM_P) + wid * 16 * SP_STRIDE;

    const __bf16* Kb = K + (size_t)b * SEQ * DKQ;
    const __bf16* Vb = VT + (size_t)b * DOUT * SEQ;
    const uint64_t kbase = (uint64_t)(uintptr_t)Kb;
    const uint64_t vbase = (uint64_t)(uintptr_t)Vb;
    // Low 32 bits of a flat LDS pointer are the raw LDS byte address
    // (ISA 10.2) -- what ASYNC_TO_LDS VDST wants.
    const uint32_t smem_lds = (uint32_t)(uintptr_t)(&smem[0]);

    // Issue one 32-key tile (K 32x64 + V^T 256x32) as per-lane async b128
    // global->LDS transfers. 10 transfers per thread, tracked by ASYNCcnt.
    auto issue_tile = [&](int j0, int bufsel) {
        const uint32_t kb = smem_lds + (uint32_t)bufsel * BUF_BYTES;
        const uint32_t vb = kb + SK_BYTES;
#pragma unroll
        for (int k = 0; k < 2; ++k) {               // K tile: 256 x b128
            const int i = (int)threadIdx.x + k * 128;
            const int row = i >> 3, c = i & 7;
            const uint32_t lds = kb + (uint32_t)(row * (SK_STRIDE * 2) + c * 16);
            const uint32_t go  = (uint32_t)((j0 + row) * (DKQ * 2) + c * 16);
            asm volatile("global_load_async_to_lds_b128 %0, %1, %2"
                         :: "v"(lds), "v"(go), "s"(kbase) : "memory");
        }
#pragma unroll
        for (int k = 0; k < 8; ++k) {               // V^T tile: 1024 x b128
            const int i = (int)threadIdx.x + k * 128;
            const int d = i >> 2, c = i & 3;
            const uint32_t lds = vb + (uint32_t)(d * (SV_STRIDE * 2) + c * 16);
            const uint32_t go  = (uint32_t)(d * (SEQ * 2) + j0 * 2 + c * 16);
            asm volatile("global_load_async_to_lds_b128 %0, %1, %2"
                         :: "v"(lds), "v"(go), "s"(vbase) : "memory");
        }
    };

    // Preload Q A-fragments for this wave's 16 rows (Kdim 64 = 2 fragments)
    const __bf16* Qb = Q + ((size_t)b * SEQ + n0) * DKQ;
    v16bf qf[2];
#pragma unroll
    for (int f = 0; f < 2; ++f) {
        const __bf16* base = Qb + (size_t)lo * DKQ + f * 32 + 8 * hi;
        union { uint4 u[2]; v16bf v; } t;
        t.u[0] = *(const uint4*)(base);        // halfs 0-7  (K = 8*hi + 0..7)
        t.u[1] = *(const uint4*)(base + 16);   // halfs 8-15 (K = 16+8*hi+0..7)
        qf[f] = t.v;
    }

    float mrun[8], lrun[8];
    v8f acc[16];
#pragma unroll
    for (int r = 0; r < 8; ++r) { mrun[r] = -3.0e38f; lrun[r] = 0.0f; }
#pragma unroll
    for (int t = 0; t < 16; ++t) { v8f z = {}; acc[t] = z; }

    issue_tile(0, 0);                                   // prime the pipeline

    for (int jt = 0; jt < SEQ / 32; ++jt) {
        asm volatile("s_wait_asynccnt 0x0" ::: "memory");  // tile jt landed
        __syncthreads();                                   // visible block-wide
        if (jt + 1 < SEQ / 32)
            issue_tile((jt + 1) * 32, (jt + 1) & 1);       // prefetch next tile

        const __bf16* sK = (const __bf16*)(smem + (jt & 1) * BUF_BYTES);
        const __bf16* sV = (const __bf16*)(smem + (jt & 1) * BUF_BYTES + SK_BYTES);

        // S = Q * K^T : two 16x16 score tiles (keys jsub*16..+15)
        v8f s[2];
#pragma unroll
        for (int js = 0; js < 2; ++js) {
            v8f c0 = {};
#pragma unroll
            for (int f = 0; f < 2; ++f) {
                // B fragment: N = key (js*16+lo), K = d = f*32 + 8*hi + 0..7, f*32 + 16 + 8*hi + 0..7
                union { uint4 u[2]; v16bf v; } tb;
                const __bf16* base = sK + (js * 16 + lo) * SK_STRIDE + f * 32 + 8 * hi;
                tb.u[0] = *(const uint4*)(base);
                tb.u[1] = *(const uint4*)(base + 16);
                c0 = __builtin_amdgcn_wmma_f32_16x16x32_bf16(false, qf[f], false, tb.v,
                                                             (short)0, c0, false, false);
            }
            s[js] = c0;
        }

        // Online softmax. Row m = r + 8*hi lives in VGPR r across one 16-lane
        // half -> butterfly over xor masks 1,2,4,8 stays inside the half.
        float p0[8], p1[8];
#pragma unroll
        for (int r = 0; r < 8; ++r) {
            float v = fmaxf(s[0][r], s[1][r]);
            v = fmaxf(v, __shfl_xor(v, 1, 32));
            v = fmaxf(v, __shfl_xor(v, 2, 32));
            v = fmaxf(v, __shfl_xor(v, 4, 32));
            v = fmaxf(v, __shfl_xor(v, 8, 32));
            const float mnew  = fmaxf(mrun[r], v);
            const float alpha = __expf(mrun[r] - mnew);
            p0[r] = __expf(s[0][r] - mnew);
            p1[r] = __expf(s[1][r] - mnew);
            float sum = p0[r] + p1[r];
            sum += __shfl_xor(sum, 1, 32);
            sum += __shfl_xor(sum, 2, 32);
            sum += __shfl_xor(sum, 4, 32);
            sum += __shfl_xor(sum, 8, 32);
            lrun[r] = lrun[r] * alpha + sum;
            mrun[r] = mnew;
#pragma unroll
            for (int t = 0; t < 16; ++t) acc[t][r] *= alpha;
        }

        // C-layout -> A-layout transpose of P through this wave's LDS tile
#pragma unroll
        for (int r = 0; r < 8; ++r) {
            const int m = r + 8 * hi;
            sP[m * SP_STRIDE + lo]      = (__bf16)p0[r];
            sP[m * SP_STRIDE + 16 + lo] = (__bf16)p1[r];
        }
        __builtin_amdgcn_wave_barrier();   // same-wave LDS ops are in order

        // P A-fragment: row M = lo, K(key) = 8*hi+0..7 and 16+8*hi+0..7
        union { uint4 u[2]; v16bf v; } tp;
        const __bf16* pb = sP + lo * SP_STRIDE + 8 * hi;
        tp.u[0] = *(const uint4*)(pb);
        tp.u[1] = *(const uint4*)(pb + 16);

        // O += P * V  (16 output-channel tiles)
#pragma unroll
        for (int t = 0; t < 16; ++t) {
            union { uint4 u[2]; v16bf v; } tv;
            const __bf16* vb = sV + (t * 16 + lo) * SV_STRIDE + 8 * hi;
            tv.u[0] = *(const uint4*)(vb);
            tv.u[1] = *(const uint4*)(vb + 16);
            acc[t] = __builtin_amdgcn_wmma_f32_16x16x32_bf16(false, tp.v, false, tv.v,
                                                             (short)0, acc[t], false, false);
        }
    }

    // Epilogue: normalize, stage via LDS, coalesced (B, 256, N) f32 stores
    float linv[8];
#pragma unroll
    for (int r = 0; r < 8; ++r) linv[r] = 1.0f / lrun[r];

    float* sO = (float*)smem;   // [64 d][68] per group of 4 d-tiles
#pragma unroll 1
    for (int g = 0; g < 4; ++g) {
        __syncthreads();
#pragma unroll
        for (int t = 0; t < 4; ++t) {
            const int dt = g * 4 + t;
#pragma unroll
            for (int r = 0; r < 8; ++r) {
                const int m = r + 8 * hi;
                sO[(t * 16 + lo) * SO_STRIDE + wid * 16 + m] = acc[dt][r] * linv[r];
            }
        }
        __syncthreads();
        for (int i = threadIdx.x; i < 64 * 16; i += 128) {
            const int dr = i >> 4, c = i & 15;
            const float4 vv = *(const float4*)(sO + dr * SO_STRIDE + c * 4);
            *(float4*)(out + ((size_t)b * DOUT + g * 64 + dr) * SEQ + n0blk + c * 4) = vv;
        }
    }
}

// ---------------------------------------------------------------------------
extern "C" void kernel_launch(void* const* d_in, const int* in_sizes, int n_in,
                              void* d_out, int out_size, void* d_ws, size_t ws_size,
                              hipStream_t stream) {
    const float* x  = (const float*)d_in[0];
    const float* Wq = (const float*)d_in[1];
    const float* Wk = (const float*)d_in[2];
    const float* Wv = (const float*)d_in[3];

    // workspace carve: Q (4MB) | K (4MB) | V^T (16MB), all bf16
    __bf16* Q  = (__bf16*)d_ws;
    __bf16* K  = Q + (size_t)BATCH * SEQ * DKQ;
    __bf16* VT = K + (size_t)BATCH * SEQ * DKQ;

    // proj: one wave per 16x16 tile: 8 batches * 256 n-tiles * 24 d-tiles waves
    const int proj_waves  = BATCH * (SEQ / 16) * 24;
    const int proj_blocks = proj_waves / 8;            // 256 threads = 8 waves
    proj_kernel<<<proj_blocks, 256, 0, stream>>>(x, Wq, Wk, Wv, Q, K, VT);

    // attention: 64 query rows per 128-thread block
    attn_kernel<<<BATCH * (SEQ / 64), 128, 0, stream>>>(Q, K, VT, (float*)d_out);
}
